// LPSA_48704929137171
// MI455X (gfx1250) — matrix operand order repaired and checked by reading, not compiled, not run on hardware
//
#include <hip/hip_runtime.h>
#include <hip/hip_bf16.h>

typedef _Float16 v16h __attribute__((ext_vector_type(16)));
typedef _Float16 v8h  __attribute__((ext_vector_type(8)));
typedef float    v8f  __attribute__((ext_vector_type(8)));

#define WMMA_F16(a, b, c) \
  __builtin_amdgcn_wmma_f32_16x16x32_f16(false, (a), false, (b), (short)0, (c), false, false)

// Build a 16-half operand from two 8-half (16B-aligned) runs.
static __device__ __forceinline__ v16h load_a16(const _Float16* p0, const _Float16* p1) {
  v8h lo = *(const v8h*)p0;
  v8h hi = *(const v8h*)p1;
  return __builtin_shufflevector(lo, hi, 0,1,2,3,4,5,6,7,8,9,10,11,12,13,14,15);
}

// ---------------------------------------------------------------------------
// K0: convert wo [256,512] and w_gate [256,256] to f16
// ---------------------------------------------------------------------------
__global__ void to_f16_kernel(const float* __restrict__ wo, const float* __restrict__ wg,
                              _Float16* __restrict__ woh, _Float16* __restrict__ wgh) {
  int i = blockIdx.x * 256 + threadIdx.x;
  if (i < 131072) woh[i] = (_Float16)wo[i];
  int j = i - 131072;
  if (j >= 0 && j < 65536) wgh[j] = (_Float16)wg[j];
}

// ---------------------------------------------------------------------------
// K1: low-rank + expansion projections. 16 tokens per block (coalesced x reads).
//   Q, K stored [head][n][64] f16 ; V stored transposed Vt[head][d][4096] f16
// ---------------------------------------------------------------------------
__global__ void proj_qkv(const float* __restrict__ x,
                         const float* __restrict__ wq1, const float* __restrict__ wk1,
                         const float* __restrict__ wv1,
                         const float* __restrict__ wq2, const float* __restrict__ wk2,
                         const float* __restrict__ wv2,
                         _Float16* __restrict__ Q, _Float16* __restrict__ Km,
                         _Float16* __restrict__ Vt) {
  __shared__ float t[256][16];    // t[c][i] = x[c][nb+i]
  __shared__ float low[96][16];   // 32 q / 32 k / 32 v low-rank rows per token
  int nb  = blockIdx.x * 16;
  int tid = threadIdx.x;
#pragma unroll
  for (int rep = 0; rep < 16; ++rep) {
    int g = rep * 256 + tid;
    int c = g >> 4, i = g & 15;
    t[c][i] = x[c * 4096 + nb + i];
  }
  __syncthreads();
  for (int task = tid; task < 1536; task += 256) {   // 96 rows x 16 tokens
    int j = task >> 4, i = task & 15;
    const float* w = (j < 32) ? (wq1 + j * 256)
                   : (j < 64) ? (wk1 + (j - 32) * 256)
                              : (wv1 + (j - 64) * 256);
    float acc = 0.f;
    for (int c = 0; c < 256; ++c) acc += t[c][i] * w[c];
    low[j][i] = acc;
  }
  __syncthreads();
  for (int task = tid; task < 8192; task += 256) {   // 512 outputs x 16 tokens
    int o = task >> 4, i = task & 15;
    int n = nb + i;
    const float* wq = wq2 + o * 32;
    const float* wk = wk2 + o * 32;
    const float* wv = wv2 + o * 32;
    float q = 0.f, k = 0.f, v = 0.f;
#pragma unroll
    for (int r = 0; r < 32; ++r) {
      q += low[r][i]      * wq[r];
      k += low[32 + r][i] * wk[r];
      v += low[64 + r][i] * wv[r];
    }
    int head = o >> 6, d = o & 63;
    Q [(head * 4096 + n) * 64 + d] = (_Float16)q;
    Km[(head * 4096 + n) * 64 + d] = (_Float16)k;
    Vt[(head * 64 + d) * 4096 + n] = (_Float16)v;
  }
}

// ---------------------------------------------------------------------------
// K2: flash attention, 64-key blocks. One wave per (head, 16-query tile).
// Row sums computed by an extra WMMA against an all-ones B operand (layout of
// D rows matches the running-stat register layout), removing all sum shuffles.
// ---------------------------------------------------------------------------
__global__ __launch_bounds__(32) void attn_kernel(const _Float16* __restrict__ Q,
                                                  const _Float16* __restrict__ Km,
                                                  const _Float16* __restrict__ Vt,
                                                  _Float16* __restrict__ AO) {
  __shared__ _Float16 P[16 * 64];        // P tile, [row m][key] layout
  int head  = blockIdx.x >> 8;
  int qbase = (blockIdx.x & 255) << 4;
  int lane  = threadIdx.x;
  int half  = lane >> 4;
  int col   = lane & 15;

  const _Float16* Qh = Q  + (head * 4096 + qbase) * 64;
  const _Float16* Kh = Km + head * 4096 * 64;
  const _Float16* Vh = Vt + head * 64 * 4096;

  // Q A-fragments (persist): row m = col, K sub-range split by half.
  v16h aq0 = load_a16(Qh + col * 64 + 8 * half,      Qh + col * 64 + 16 + 8 * half);
  v16h aq1 = load_a16(Qh + col * 64 + 32 + 8 * half, Qh + col * 64 + 48 + 8 * half);

  v16h vones;
#pragma unroll
  for (int e = 0; e < 16; ++e) vones[e] = (_Float16)1.0f;

  v8f o0 = {}, o1 = {}, o2 = {}, o3 = {};
  v8f rs = {};                           // running row-sum (all cols identical)
  float rmax[8];
#pragma unroll
  for (int r = 0; r < 8; ++r) rmax[r] = -3.0e38f;

  for (int kb = 0; kb < 4096; kb += 64) {
    // S = Q.K^T for 4 key tiles (B col = key, elements = d, contiguous loads)
    v8f z = {};
    v8f s0, s1, s2, s3;
    {
      const _Float16* k0 = Kh + (kb +  0 + col) * 64;
      const _Float16* k1 = Kh + (kb + 16 + col) * 64;
      const _Float16* k2 = Kh + (kb + 32 + col) * 64;
      const _Float16* k3 = Kh + (kb + 48 + col) * 64;
      s0 = WMMA_F16(aq0, load_a16(k0 + 8 * half, k0 + 16 + 8 * half), z);
      s0 = WMMA_F16(aq1, load_a16(k0 + 32 + 8 * half, k0 + 48 + 8 * half), s0);
      s1 = WMMA_F16(aq0, load_a16(k1 + 8 * half, k1 + 16 + 8 * half), z);
      s1 = WMMA_F16(aq1, load_a16(k1 + 32 + 8 * half, k1 + 48 + 8 * half), s1);
      s2 = WMMA_F16(aq0, load_a16(k2 + 8 * half, k2 + 16 + 8 * half), z);
      s2 = WMMA_F16(aq1, load_a16(k2 + 32 + 8 * half, k2 + 48 + 8 * half), s2);
      s3 = WMMA_F16(aq0, load_a16(k3 + 8 * half, k3 + 16 + 8 * half), z);
      s3 = WMMA_F16(aq1, load_a16(k3 + 32 + 8 * half, k3 + 48 + 8 * half), s3);
    }
    // WGP-scope prefetch of the next K block (32 lanes x 256B stride = 8KB)
    __builtin_prefetch(Kh + (kb + 64) * 64 + lane * 128, 0, 3);

    // Online softmax over 64 new keys: one 4-step max butterfly per row.
#pragma unroll
    for (int r = 0; r < 8; ++r) {
      float a0 = s0[r] * 0.125f;         // 1/sqrt(64)
      float a1 = s1[r] * 0.125f;
      float a2 = s2[r] * 0.125f;
      float a3 = s3[r] * 0.125f;
      float mx = fmaxf(fmaxf(a0, a1), fmaxf(a2, a3));
      mx = fmaxf(mx, __shfl_xor(mx, 1, 32));
      mx = fmaxf(mx, __shfl_xor(mx, 2, 32));
      mx = fmaxf(mx, __shfl_xor(mx, 4, 32));
      mx = fmaxf(mx, __shfl_xor(mx, 8, 32));
      float mnew  = fmaxf(rmax[r], mx);
      float alpha = __expf(rmax[r] - mnew);
      rmax[r] = mnew;
      float p0 = __expf(a0 - mnew);
      float p1 = __expf(a1 - mnew);
      float p2 = __expf(a2 - mnew);
      float p3 = __expf(a3 - mnew);
      rs[r] *= alpha;
      o0[r] *= alpha; o1[r] *= alpha; o2[r] *= alpha; o3[r] *= alpha;
      int row = r + 8 * half;
      P[row * 64 +  0 + col] = (_Float16)p0;
      P[row * 64 + 16 + col] = (_Float16)p1;
      P[row * 64 + 32 + col] = (_Float16)p2;
      P[row * 64 + 48 + col] = (_Float16)p3;
    }
    __syncthreads();                     // single wave: ordering only

    // P as A-operands (row m = col): keys 0..31 and 32..63 of this block.
    v16h ap0 = load_a16(&P[col * 64 + 8 * half],      &P[col * 64 + 16 + 8 * half]);
    v16h ap1 = load_a16(&P[col * 64 + 32 + 8 * half], &P[col * 64 + 48 + 8 * half]);

    // Row sums via WMMA: D = P . ones  (every column of D = row sum)
    rs = WMMA_F16(ap0, vones, rs);
    rs = WMMA_F16(ap1, vones, rs);

    // O += P.V ; Vt B-fragments are contiguous per lane (d-major layout).
#pragma unroll
    for (int dt = 0; dt < 4; ++dt) {
      const _Float16* vp = Vh + (dt * 16 + col) * 4096 + kb;
      v16h bv0 = load_a16(vp + 8 * half,      vp + 16 + 8 * half);
      v16h bv1 = load_a16(vp + 32 + 8 * half, vp + 48 + 8 * half);
      v8f* o = (dt == 0) ? &o0 : (dt == 1) ? &o1 : (dt == 2) ? &o2 : &o3;
      *o = WMMA_F16(ap0, bv0, *o);
      *o = WMMA_F16(ap1, bv1, *o);
    }
    __syncthreads();
  }

#pragma unroll
  for (int r = 0; r < 8; ++r) {
    float inv = 1.0f / rs[r];
    int row = qbase + r + 8 * half;
    _Float16* dst = AO + row * 512 + head * 64;
    dst[0 * 16 + col] = (_Float16)(o0[r] * inv);
    dst[1 * 16 + col] = (_Float16)(o1[r] * inv);
    dst[2 * 16 + col] = (_Float16)(o2[r] * inv);
    dst[3 * 16 + col] = (_Float16)(o3[r] * inv);
  }
}

// ---------------------------------------------------------------------------
// K3: output projection y[c][n] = AO[n][:] . wo[c][:] + bo[c]  (WMMA, K=512)
// ---------------------------------------------------------------------------
__global__ __launch_bounds__(32) void out_proj(const _Float16* __restrict__ AO,
                                               const _Float16* __restrict__ WOh,
                                               const float* __restrict__ bo,
                                               float* __restrict__ Y) {
  int nt = blockIdx.x >> 4, ct = blockIdx.x & 15;
  int nbase = nt * 16, cbase = ct * 16;
  int lane = threadIdx.x, half = lane >> 4, col = lane & 15;
  v8f acc = {};
  const _Float16* arow = AO  + (nbase + col) * 512;   // A row = token
  const _Float16* brow = WOh + (cbase + col) * 512;   // B col = out channel
#pragma unroll
  for (int ch = 0; ch < 16; ++ch) {
    v16h a = load_a16(arow + 32 * ch + 8 * half,  arow + 32 * ch + 16 + 8 * half);
    v16h b = load_a16(brow + 32 * ch + 8 * half,  brow + 32 * ch + 16 + 8 * half);
    acc = WMMA_F16(a, b, acc);
  }
  float bias = bo[cbase + col];
#pragma unroll
  for (int r = 0; r < 8; ++r) {
    Y[(cbase + col) * 4096 + nbase + r + 8 * half] = acc[r] + bias;  // CHW layout
  }
}

// ---------------------------------------------------------------------------
// K4: depthwise 3x3 conv (SAME) + bias. Emits f32 CHW result U plus a
// token-major f16 transpose Ut[n][c] so gate B-fragments are contiguous.
// ---------------------------------------------------------------------------
__global__ void dwconv(const float* __restrict__ Y, const float* __restrict__ wdw,
                       const float* __restrict__ bdw, float* __restrict__ U,
                       _Float16* __restrict__ Ut) {
  int idx = blockIdx.x * 256 + threadIdx.x;
  int c = idx >> 12, hw = idx & 4095, h = hw >> 6, w = hw & 63;
  float acc = bdw[c];
#pragma unroll
  for (int dy = 0; dy < 3; ++dy) {
    int yy = h + dy - 1;
    if (yy < 0 || yy > 63) continue;
#pragma unroll
    for (int dx = 0; dx < 3; ++dx) {
      int xx = w + dx - 1;
      if (xx < 0 || xx > 63) continue;
      acc += Y[c * 4096 + yy * 64 + xx] * wdw[c * 9 + dy * 3 + dx];
    }
  }
  U[idx]            = acc;
  Ut[hw * 256 + c]  = (_Float16)acc;
}

// ---------------------------------------------------------------------------
// K5: gate = sigmoid(Wg.U + bg), out = U * gate.  WMMA over K=256.
// A = w_gate rows (contiguous), B = Ut rows (contiguous after transpose).
// ---------------------------------------------------------------------------
__global__ __launch_bounds__(32) void gate_kernel(const float* __restrict__ U,
                                                  const _Float16* __restrict__ Ut,
                                                  const _Float16* __restrict__ WGh,
                                                  const float* __restrict__ bg,
                                                  float* __restrict__ out) {
  int nt = blockIdx.x >> 4, ct = blockIdx.x & 15;
  int nbase = nt * 16, cbase = ct * 16;
  int lane = threadIdx.x, half = lane >> 4, col = lane & 15;
  v8f acc = {};
  const _Float16* arow = WGh + (cbase + col) * 256;   // A row = out channel
  const _Float16* brow = Ut  + (nbase + col) * 256;   // B col = token
#pragma unroll
  for (int ch = 0; ch < 8; ++ch) {
    v16h a = load_a16(arow + 32 * ch + 8 * half,  arow + 32 * ch + 16 + 8 * half);
    v16h b = load_a16(brow + 32 * ch + 8 * half,  brow + 32 * ch + 16 + 8 * half);
    acc = WMMA_F16(a, b, acc);
  }
#pragma unroll
  for (int r = 0; r < 8; ++r) {
    int c = cbase + r + 8 * half;
    int n = nbase + col;
    float g = acc[r] + bg[c];
    g = 1.0f / (1.0f + __expf(-g));
    out[c * 4096 + n] = U[c * 4096 + n] * g;
  }
}

// ---------------------------------------------------------------------------
extern "C" void kernel_launch(void* const* d_in, const int* in_sizes, int n_in,
                              void* d_out, int out_size, void* d_ws, size_t ws_size,
                              hipStream_t stream) {
  (void)in_sizes; (void)n_in; (void)out_size; (void)ws_size;
  const float* x   = (const float*)d_in[0];
  const float* wq1 = (const float*)d_in[1];
  const float* wk1 = (const float*)d_in[2];
  const float* wv1 = (const float*)d_in[3];
  const float* wq2 = (const float*)d_in[4];
  const float* wk2 = (const float*)d_in[5];
  const float* wv2 = (const float*)d_in[6];
  const float* wo  = (const float*)d_in[7];
  const float* bo  = (const float*)d_in[8];
  const float* wdw = (const float*)d_in[9];
  const float* bdw = (const float*)d_in[10];
  const float* wg  = (const float*)d_in[11];
  const float* bg  = (const float*)d_in[12];

  char* ws = (char*)d_ws;
  _Float16* Q   = (_Float16*)(ws + (size_t)(0)  * (1 << 20));
  _Float16* Km  = (_Float16*)(ws + (size_t)(4)  * (1 << 20));
  _Float16* Vt  = (_Float16*)(ws + (size_t)(8)  * (1 << 20));
  _Float16* AO  = (_Float16*)(ws + (size_t)(12) * (1 << 20));
  float*    Y   = (float*)   (ws + (size_t)(16) * (1 << 20));
  float*    U   = (float*)   (ws + (size_t)(20) * (1 << 20));
  _Float16* Ut  = (_Float16*)(ws + (size_t)(24) * (1 << 20));
  _Float16* WOh = (_Float16*)(ws + (size_t)(26) * (1 << 20));
  _Float16* WGh = (_Float16*)(ws + (size_t)(26) * (1 << 20) + 262144);

  to_f16_kernel<<<768, 256, 0, stream>>>(wo, wg, WOh, WGh);
  proj_qkv<<<256, 256, 0, stream>>>(x, wq1, wk1, wv1, wq2, wk2, wv2, Q, Km, Vt);
  attn_kernel<<<2048, 32, 0, stream>>>(Q, Km, Vt, AO);
  out_proj<<<4096, 32, 0, stream>>>(AO, WOh, bo, Y);
  dwconv<<<4096, 256, 0, stream>>>(Y, wdw, bdw, U, Ut);
  gate_kernel<<<4096, 32, 0, stream>>>(U, Ut, WGh, bg, (float*)d_out);
}
